// MultiHeadAttention_89653147337364
// MI455X (gfx1250) — matrix operand order repaired and checked by reading, not compiled, not run on hardware
//
#include <hip/hip_runtime.h>
#include <hip/hip_bf16.h>

typedef __bf16 v8bf  __attribute__((ext_vector_type(8)));
typedef __bf16 v16bf __attribute__((ext_vector_type(16)));
typedef float  v8f   __attribute__((ext_vector_type(8)));
typedef unsigned int u32x4 __attribute__((ext_vector_type(4)));
typedef int          i32x8 __attribute__((ext_vector_type(8)));
typedef int          i32x4 __attribute__((ext_vector_type(4)));

union BF16x16 { v16bf v; v8bf h[2]; };

#define D_MODEL 512
#define NH      8
#define DK      64
#define SEQ     4096
#define BATCH   2
#define MROWS   (BATCH * SEQ)   // 8192

// ---------------------------------------------------------------------------
// fp32 -> bf16 elementwise convert
// ---------------------------------------------------------------------------
__global__ void cvt_bf16_kernel(const float* __restrict__ x,
                                __bf16* __restrict__ y, int n) {
    int i = blockIdx.x * blockDim.x + threadIdx.x;
    int stride = gridDim.x * blockDim.x;
    for (; i < n; i += stride) y[i] = (__bf16)x[i];
}

// ---------------------------------------------------------------------------
// TDM: issue a 2D tensor_load_to_lds. Tile = tile_d1 rows x tile_d0 elements
// (bf16, 2 bytes), global row stride = stride_d0 elements, packed into LDS
// row-major. D# bitfields per CDNA5 ISA 8.3/8.4 (group0: count/lds/global/type,
// group1: data_size, tensor dims == tile dims, dim0 stride). Groups 2/3 = 0.
// This toolchain exposes the 6-arg builtin (g0, g1, g2, g3, g4, cpol).
// ---------------------------------------------------------------------------
__device__ __forceinline__ void tdm_load_2d(unsigned int lds_addr,
                                            const void* gptr,
                                            unsigned int tile_d0,
                                            unsigned int tile_d1,
                                            unsigned int stride_d0) {
    unsigned long long ga = (unsigned long long)(uintptr_t)gptr;
    u32x4 g0;
    g0[0] = 1u;                                   // count=1 (valid), user mode
    g0[1] = lds_addr;                             // LDS byte offset
    g0[2] = (unsigned int)ga;                     // global_addr[31:0]
    g0[3] = (unsigned int)((ga >> 32) & 0x1FFFFFFu) | (2u << 30);  // type=2
    i32x8 g1;
    g1[0] = (int)(1u << 16);                      // data_size=1 -> 2 bytes
    g1[1] = (int)((tile_d0 & 0xFFFFu) << 16);     // tensor_dim0[15:0]
    g1[2] = (int)(((tile_d0 >> 16) & 0xFFFFu) |
                  ((tile_d1 & 0xFFFFu) << 16));   // t_dim0[31:16] | t_dim1[15:0]
    g1[3] = (int)(((tile_d1 >> 16) & 0xFFFFu) |
                  (tile_d0 << 16));               // t_dim1[31:16] | tile_dim0
    g1[4] = (int)(tile_d1 & 0xFFFFu);             // tile_dim1 | tile_dim2=0
    g1[5] = (int)stride_d0;                       // tensor_dim0_stride[31:0]
    g1[6] = 0;                                    // stride hi | dim1_stride lo
    g1[7] = 0;
    i32x4 z4 = {0, 0, 0, 0};
    i32x8 z8 = {0, 0, 0, 0, 0, 0, 0, 0};
    __builtin_amdgcn_tensor_load_to_lds(g0, g1, z4, z4, z8, 0);
}

__device__ __forceinline__ unsigned int lds_off(const void* p) {
    // flat->LDS aperture mapping uses addr[31:0] as the LDS byte address
    return (unsigned int)(uintptr_t)p;
}

// ---------------------------------------------------------------------------
// Shared GEMM core: C(128x64 per block) = A(row-major MxK) * BT(row-major NxK)^T
// 256 threads = 8 waves; each wave owns a 32x32 tile = 2x2 WMMA tiles.
// ---------------------------------------------------------------------------
__device__ __forceinline__ void gemm_block(const __bf16* __restrict__ A,
                                           const __bf16* __restrict__ BT,
                                           __bf16* sA, __bf16* sB,
                                           int bm0, int bn0, v8f acc[2][2]) {
    const int tid  = threadIdx.x;
    const int lane = tid & 31, wid = tid >> 5;
    const int wm = (wid & 3) * 32, wn = (wid >> 2) * 32;
    const int ln = lane & 15, hl = lane >> 4;

    for (int kc = 0; kc < D_MODEL; kc += 32) {
#pragma unroll
        for (int i = 0; i < 2; ++i) {
            int e = (tid + i * 256) * 8;
            int r = e >> 5, c = e & 31;
            *(v8bf*)&sA[e] = *(const v8bf*)&A[(bm0 + r) * D_MODEL + kc + c];
        }
        {
            int e = tid * 8;
            int r = e >> 5, c = e & 31;
            *(v8bf*)&sB[e] = *(const v8bf*)&BT[(bn0 + r) * D_MODEL + kc + c];
        }
        __syncthreads();

        BF16x16 afr[2];
        v16bf   bfr[2];
#pragma unroll
        for (int mt = 0; mt < 2; ++mt) {
            int r = wm + mt * 16 + ln;
            afr[mt].h[0] = *(const v8bf*)&sA[r * 32 + 16 * hl];
            afr[mt].h[1] = *(const v8bf*)&sA[r * 32 + 16 * hl + 8];
        }
#pragma unroll
        for (int nt = 0; nt < 2; ++nt) {
            int r = wn + nt * 16 + ln;
            bfr[nt] = *(const v16bf*)&sB[r * 32 + 16 * hl];
        }
#pragma unroll
        for (int mt = 0; mt < 2; ++mt)
#pragma unroll
            for (int nt = 0; nt < 2; ++nt)
                acc[mt][nt] = __builtin_amdgcn_wmma_f32_16x16x32_bf16(
                    false, afr[mt].v, false, bfr[nt], (short)0, acc[mt][nt],
                    false, false);
        __syncthreads();
    }
}

// ---------------------------------------------------------------------------
// Projection: Y = X @ W^T + b, written into head layout (bf16).
// vtrans=0 : dst[((b*NH+h)*SEQ + s)*DK + dk]   (Q, K)
// vtrans=1 : dst[((b*NH+h)*DK + dk)*SEQ + s]   (V, transposed for P*V)
// ---------------------------------------------------------------------------
__global__ __launch_bounds__(256) void proj_heads_kernel(
    const __bf16* __restrict__ A, const __bf16* __restrict__ BT,
    const float* __restrict__ bias, __bf16* __restrict__ dst, int vtrans) {
    __shared__ alignas(64) __bf16 sA[128 * 32];
    __shared__ alignas(64) __bf16 sB[64 * 32];
    v8f acc[2][2] = {};
    const int bm0 = blockIdx.x * 128, bn0 = blockIdx.y * 64;
    gemm_block(A, BT, sA, sB, bm0, bn0, acc);

    const int lane = threadIdx.x & 31, wid = threadIdx.x >> 5;
    const int wm = (wid & 3) * 32, wn = (wid >> 2) * 32;
    const int ln = lane & 15, hl = lane >> 4;
#pragma unroll
    for (int mt = 0; mt < 2; ++mt)
#pragma unroll
        for (int nt = 0; nt < 2; ++nt) {
            int gn = bn0 + wn + nt * 16 + ln;
            float bv = bias[gn];
            int h = gn >> 6, dk = gn & 63;
#pragma unroll
            for (int vv = 0; vv < 8; ++vv) {
                int gm = bm0 + wm + mt * 16 + vv + 8 * hl;
                int b = gm >> 12, s = gm & 4095;
                float val = acc[mt][nt][vv] + bv;
                if (!vtrans)
                    dst[((b * NH + h) * SEQ + s) * DK + dk] = (__bf16)val;
                else
                    dst[((b * NH + h) * DK + dk) * SEQ + s] = (__bf16)val;
            }
        }
}

// ---------------------------------------------------------------------------
// Output projection: out = X @ Wo^T + bo (fp32 result)
// ---------------------------------------------------------------------------
__global__ __launch_bounds__(256) void out_proj_kernel(
    const __bf16* __restrict__ A, const __bf16* __restrict__ BT,
    const float* __restrict__ bias, float* __restrict__ dst) {
    __shared__ alignas(64) __bf16 sA[128 * 32];
    __shared__ alignas(64) __bf16 sB[64 * 32];
    v8f acc[2][2] = {};
    const int bm0 = blockIdx.x * 128, bn0 = blockIdx.y * 64;
    gemm_block(A, BT, sA, sB, bm0, bn0, acc);

    const int lane = threadIdx.x & 31, wid = threadIdx.x >> 5;
    const int wm = (wid & 3) * 32, wn = (wid >> 2) * 32;
    const int ln = lane & 15, hl = lane >> 4;
#pragma unroll
    for (int mt = 0; mt < 2; ++mt)
#pragma unroll
        for (int nt = 0; nt < 2; ++nt) {
            int gn = bn0 + wn + nt * 16 + ln;
            float bv = bias[gn];
#pragma unroll
            for (int vv = 0; vv < 8; ++vv) {
                int gm = bm0 + wm + mt * 16 + vv + 8 * hl;
                dst[gm * D_MODEL + gn] = acc[mt][nt][vv] + bv;
            }
        }
}

// ---------------------------------------------------------------------------
// Flash attention: each wave owns 16 query rows; online softmax in fp32;
// K/V chunks staged in LDS by the Tensor Data Mover (double-buffered,
// TENSORcnt-synchronized), shared by all 8 waves of the block.
// Kh: [B,H,S,DK] row-major; Vt: [B,H,DK,S] row-major.
// ---------------------------------------------------------------------------
__global__ __launch_bounds__(256) void flash_attn_kernel(
    const __bf16* __restrict__ Qh, const __bf16* __restrict__ Kh,
    const __bf16* __restrict__ Vt, __bf16* __restrict__ Xm) {
    __shared__ alignas(128) __bf16 sK[2][32 * DK];   // [key=32][dk=64]
    __shared__ alignas(128) __bf16 sV[2][DK * 32];   // [dk=64][key=32]
    __shared__ alignas(64)  __bf16 P[8][16 * 32];    // per-wave P tile
    const int tid = threadIdx.x, lane = tid & 31, wid = tid >> 5;
    const int ln = lane & 15, hl = lane >> 4;
    const int bh = blockIdx.y;                // b*NH + h
    const int b = bh >> 3, h = bh & 7;
    const int q0 = blockIdx.x * 128 + wid * 16;

    const __bf16* Qb = Qh + (size_t)bh * SEQ * DK;
    const __bf16* Kb = Kh + (size_t)bh * SEQ * DK;
    const __bf16* Vb = Vt + (size_t)bh * DK * SEQ;
    __bf16* pl = &P[wid][0];

    // Q A-fragments (16x64), loaded once, reused for all key chunks
    BF16x16 qa[2];
#pragma unroll
    for (int kp = 0; kp < 2; ++kp) {
        const __bf16* qr = Qb + (q0 + ln) * DK + kp * 32;
        qa[kp].h[0] = *(const v8bf*)&qr[16 * hl];
        qa[kp].h[1] = *(const v8bf*)&qr[16 * hl + 8];
    }

    v8f Ot[4] = {};
    float mrow[8], lrow[8];
#pragma unroll
    for (int vv = 0; vv < 8; ++vv) { mrow[vv] = -3.0e38f; lrow[vv] = 0.0f; }

    // prologue: TDM chunk 0 into buffer 0 (wave 0 only; EXEC ignored by TDM)
    if (wid == 0) {
        tdm_load_2d(lds_off(&sK[0][0]), Kb, DK, 32, DK);      // 32 rows x 64
        tdm_load_2d(lds_off(&sV[0][0]), Vb, 32, DK, SEQ);     // 64 rows x 32
    }

    for (int kc = 0; kc < SEQ; kc += 32) {
        const int cur = (kc >> 5) & 1, nxt = cur ^ 1;
        if (wid == 0) {
            if (kc + 32 < SEQ) {
                tdm_load_2d(lds_off(&sK[nxt][0]), Kb + (kc + 32) * DK, DK, 32, DK);
                tdm_load_2d(lds_off(&sV[nxt][0]), Vb + (kc + 32), 32, DK, SEQ);
                __builtin_amdgcn_s_wait_tensorcnt(2);  // chunk kc landed
            } else {
                __builtin_amdgcn_s_wait_tensorcnt(0);
            }
        }
        __syncthreads();

        // scores: S(16x32) = Q(16x64) * K^T, two 16x16 C tiles (from LDS)
        v8f sc[2];
#pragma unroll
        for (int nt = 0; nt < 2; ++nt) {
            const __bf16* kr = &sK[cur][(nt * 16 + ln) * DK];
            v16bf kb0 = *(const v16bf*)&kr[16 * hl];
            v16bf kb1 = *(const v16bf*)&kr[32 + 16 * hl];
            v8f z = {};
            z = __builtin_amdgcn_wmma_f32_16x16x32_bf16(
                false, qa[0].v, false, kb0, (short)0, z, false, false);
            z = __builtin_amdgcn_wmma_f32_16x16x32_bf16(
                false, qa[1].v, false, kb1, (short)0, z, false, false);
            sc[nt] = z;
        }

        // online softmax (row stats across 16 lanes of each half)
        float corr[8];
#pragma unroll
        for (int vv = 0; vv < 8; ++vv) {
            float a = sc[0][vv] * 0.125f;   // 1/sqrt(64)
            float c = sc[1][vv] * 0.125f;
            float x = fmaxf(a, c);
#pragma unroll
            for (int msk = 1; msk < 16; msk <<= 1)
                x = fmaxf(x, __shfl_xor(x, msk, 32));
            float mn = fmaxf(mrow[vv], x);
            float cr = __expf(mrow[vv] - mn);
            float p0 = __expf(a - mn);
            float p1 = __expf(c - mn);
            float rs = p0 + p1;
#pragma unroll
            for (int msk = 1; msk < 16; msk <<= 1)
                rs += __shfl_xor(rs, msk, 32);
            lrow[vv] = lrow[vv] * cr + rs;
            mrow[vv] = mn;
            corr[vv] = cr;
            // spill P in C layout; reload in A layout (lane transpose via LDS,
            // DS ops are in-order within a wave)
            pl[(vv + 8 * hl) * 32 + ln]      = (__bf16)p0;
            pl[(vv + 8 * hl) * 32 + 16 + ln] = (__bf16)p1;
        }
        __builtin_amdgcn_wave_barrier();

        BF16x16 pf;
        pf.h[0] = *(const v8bf*)&pl[ln * 32 + 16 * hl];
        pf.h[1] = *(const v8bf*)&pl[ln * 32 + 16 * hl + 8];

        // O = O*corr + P(16x32) * V(32x64)  (V from LDS)
#pragma unroll
        for (int nt = 0; nt < 4; ++nt) {
#pragma unroll
            for (int vv = 0; vv < 8; ++vv) Ot[nt][vv] *= corr[vv];
            const __bf16* vr = &sV[cur][(nt * 16 + ln) * 32];
            v16bf vb = *(const v16bf*)&vr[16 * hl];
            Ot[nt] = __builtin_amdgcn_wmma_f32_16x16x32_bf16(
                false, pf.v, false, vb, (short)0, Ot[nt], false, false);
        }
        __syncthreads();   // all waves done with buffer `cur` before reuse
    }

    // normalize and store merged-head activations (bf16)
    float rcp[8];
#pragma unroll
    for (int vv = 0; vv < 8; ++vv) rcp[vv] = 1.0f / lrow[vv];
#pragma unroll
    for (int nt = 0; nt < 4; ++nt)
#pragma unroll
        for (int vv = 0; vv < 8; ++vv) {
            int s = q0 + vv + 8 * hl;
            Xm[((size_t)(b * SEQ + s)) * D_MODEL + h * DK + nt * 16 + ln] =
                (__bf16)(Ot[nt][vv] * rcp[vv]);
        }
}

// ---------------------------------------------------------------------------
// Host-side orchestration
// ---------------------------------------------------------------------------
extern "C" void kernel_launch(void* const* d_in, const int* in_sizes, int n_in,
                              void* d_out, int out_size, void* d_ws,
                              size_t ws_size, hipStream_t stream) {
    (void)in_sizes; (void)n_in; (void)out_size; (void)ws_size;
    const float* q  = (const float*)d_in[0];
    const float* k  = (const float*)d_in[1];
    const float* v  = (const float*)d_in[2];
    const float* wq = (const float*)d_in[3];
    const float* bq = (const float*)d_in[4];
    const float* wk = (const float*)d_in[5];
    const float* bk = (const float*)d_in[6];
    const float* wv = (const float*)d_in[7];
    const float* bv = (const float*)d_in[8];
    const float* wo = (const float*)d_in[9];
    const float* bo = (const float*)d_in[10];

    const size_t SD = (size_t)MROWS * D_MODEL;    // 8192*512
    const size_t WW = (size_t)D_MODEL * D_MODEL;  // 512*512

    char* ws = (char*)d_ws;
    __bf16* qbf  = (__bf16*)ws; ws += SD * 2;
    __bf16* kbf  = (__bf16*)ws; ws += SD * 2;
    __bf16* vbf  = (__bf16*)ws; ws += SD * 2;
    __bf16* wqbf = (__bf16*)ws; ws += WW * 2;
    __bf16* wkbf = (__bf16*)ws; ws += WW * 2;
    __bf16* wvbf = (__bf16*)ws; ws += WW * 2;
    __bf16* wobf = (__bf16*)ws; ws += WW * 2;
    __bf16* Qhb  = (__bf16*)ws; ws += SD * 2;
    __bf16* Khb  = (__bf16*)ws; ws += SD * 2;
    __bf16* Vtb  = (__bf16*)ws; ws += SD * 2;
    __bf16* Xmb  = (__bf16*)ws; ws += SD * 2;

    cvt_bf16_kernel<<<2048, 256, 0, stream>>>(q,  qbf,  (int)SD);
    cvt_bf16_kernel<<<2048, 256, 0, stream>>>(k,  kbf,  (int)SD);
    cvt_bf16_kernel<<<2048, 256, 0, stream>>>(v,  vbf,  (int)SD);
    cvt_bf16_kernel<<<256,  256, 0, stream>>>(wq, wqbf, (int)WW);
    cvt_bf16_kernel<<<256,  256, 0, stream>>>(wk, wkbf, (int)WW);
    cvt_bf16_kernel<<<256,  256, 0, stream>>>(wv, wvbf, (int)WW);
    cvt_bf16_kernel<<<256,  256, 0, stream>>>(wo, wobf, (int)WW);

    dim3 gp(MROWS / 128, D_MODEL / 64);   // (64, 8)
    proj_heads_kernel<<<gp, 256, 0, stream>>>(qbf, wqbf, bq, Qhb, 0);
    proj_heads_kernel<<<gp, 256, 0, stream>>>(kbf, wkbf, bk, Khb, 0);
    proj_heads_kernel<<<gp, 256, 0, stream>>>(vbf, wvbf, bv, Vtb, 1);

    flash_attn_kernel<<<dim3(SEQ / 128, BATCH * NH), 256, 0, stream>>>(
        Qhb, Khb, Vtb, Xmb);

    out_proj_kernel<<<gp, 256, 0, stream>>>(Xmb, wobf, bo, (float*)d_out);
}
